// TriangleMultiplication_23416161698173
// MI455X (gfx1250) — compile-verified, matrix operand order checked by reading, not run on hardware
//
#include <hip/hip_runtime.h>

// ---------------------------------------------------------------------------
// Types
// ---------------------------------------------------------------------------
typedef __attribute__((ext_vector_type(16))) __bf16 v16bf;
typedef __attribute__((ext_vector_type(8)))  __bf16 v8bf;
typedef __attribute__((ext_vector_type(8)))  float  v8f;
typedef __attribute__((ext_vector_type(4)))  float  v4f;

static constexpr int  LSEQ = 384;
static constexpr int  C    = 128;
static constexpr long P    = (long)LSEQ * LSEQ;   // 147456 positions

// ---------------------------------------------------------------------------
// Helpers
// ---------------------------------------------------------------------------
__device__ inline __bf16 f2bf(float f) {
    unsigned u = __builtin_bit_cast(unsigned, f);
    unsigned r = (u + 0x7FFFu + ((u >> 16) & 1u)) >> 16;   // RNE
    unsigned short s = (unsigned short)r;
    return __builtin_bit_cast(__bf16, s);
}

__device__ inline float bf2f(__bf16 b) {
    unsigned u = (unsigned)__builtin_bit_cast(unsigned short, b) << 16;
    return __builtin_bit_cast(float, u);
}

__device__ inline unsigned pack2(__bf16 a, __bf16 b) {
    unsigned short ua = __builtin_bit_cast(unsigned short, a);
    unsigned short ub = __builtin_bit_cast(unsigned short, b);
    return (unsigned)ua | ((unsigned)ub << 16);
}

__device__ inline v8f zero8() {
    v8f z;
#pragma unroll
    for (int h = 0; h < 8; ++h) z[h] = 0.0f;
    return z;
}

__device__ inline float sigmoidf(float x) {
    return 1.0f / (1.0f + __expf(-x));
}

__device__ inline v8f wmma_bf16(v16bf a, v16bf b, v8f c) {
    return __builtin_amdgcn_wmma_f32_16x16x32_bf16(
        /*neg_a=*/false, a, /*neg_b=*/false, b,
        /*c_mod=*/(short)0, c, /*reuse_a=*/false, /*reuse_b=*/false);
}

// Assemble a 16x32 fragment from a per-lane pointer (two 16-byte chunks at
// +0 and +16 elements).  Works for global or LDS-derived pointers.
__device__ inline v16bf frag_ptr_bf16(const __bf16* p) {
    v8bf lo = *reinterpret_cast<const v8bf*>(p);
    v8bf hi = *reinterpret_cast<const v8bf*>(p + 16);
    v16bf out;
#pragma unroll
    for (int h = 0; h < 8; ++h) { out[h] = lo[h]; out[8 + h] = hi[h]; }
    return out;
}

// 16x32 bf16 A fragment (or 32x16 B fragment of a transposed source) from a
// row-major bf16 matrix.  ISA lane layout:
//   lanes 0-15 : row = lane,     K chunks at kb and kb+16
//   lanes 16-31: row = lane-16,  K base + 8
__device__ inline v16bf frag_bf16(const __bf16* __restrict__ base, int ld,
                                  long row0, int k0, int lane) {
    long r  = row0 + (lane & 15);
    int  kb = k0 + ((lane >> 4) << 3);
    return frag_ptr_bf16(base + r * (long)ld + kb);
}

// Same fragment but converting from a row-major fp32 source.
__device__ inline v16bf frag_f32(const float* __restrict__ base, int ld,
                                 long row0, int k0, int lane) {
    long r  = row0 + (lane & 15);
    int  kb = k0 + ((lane >> 4) << 3);
    const float* p = base + r * (long)ld + kb;
    const v4f* q = reinterpret_cast<const v4f*>(p);
    v4f x0 = q[0], x1 = q[1], x2 = q[4], x3 = q[5];   // +16 floats = +4 v4f
    v16bf out;
#pragma unroll
    for (int h = 0; h < 4; ++h) {
        out[h]      = f2bf(x0[h]);
        out[4 + h]  = f2bf(x1[h]);
        out[8 + h]  = f2bf(x2[h]);
        out[12 + h] = f2bf(x3[h]);
    }
    return out;
}

// ---------------------------------------------------------------------------
// Kernel 0: fp32 -> bf16 weight conversion (6 matrices of 128x128)
// order: 0=lp_w 1=rp_w 2=lg_w 3=rg_w 4=og_w 5=ow
// ---------------------------------------------------------------------------
__global__ void k_cvt_w(const float* __restrict__ w0, const float* __restrict__ w1,
                        const float* __restrict__ w2, const float* __restrict__ w3,
                        const float* __restrict__ w4, const float* __restrict__ w5,
                        __bf16* __restrict__ dst) {
    int idx = blockIdx.x * 256 + threadIdx.x;      // < 6*16384
    int m = idx >> 14, e = idx & 16383;
    const float* s = w0;
    if (m == 1) s = w1; else if (m == 2) s = w2; else if (m == 3) s = w3;
    else if (m == 4) s = w4; else if (m == 5) s = w5;
    dst[idx] = f2bf(s[e]);
}

// ---------------------------------------------------------------------------
// GEMM phase helper: acc[nt] += A(ks) x W(nt,ks) over K=128, with all 8
// B-fragments batched per K-step so loads clause-issue ahead of the WMMAs.
// ---------------------------------------------------------------------------
__device__ inline void gemm128_lds_a(const __bf16* aBase,          // LDS, stride 136
                                     const __bf16* __restrict__ wmat,
                                     int lane, v8f acc[8]) {
#pragma unroll
    for (int ks = 0; ks < 4; ++ks) {
        v16bf a = frag_ptr_bf16(aBase + ks * 32);
        v16bf b[8];
#pragma unroll
        for (int nt = 0; nt < 8; ++nt)
            b[nt] = frag_bf16(wmat, C, nt * 16, ks * 32, lane);
#pragma unroll
        for (int nt = 0; nt < 8; ++nt)
            acc[nt] = wmma_bf16(a, b[nt], acc[nt]);
    }
}

// ---------------------------------------------------------------------------
// Kernel 1: fused proj + LN + mask + sigmoid-gate, two passes (left/right).
// Output bf16 channel-major  dst[c][p],  p = i*L + k.
// Block: 256 threads (8 waves), 128 rows of p, full C=128 outputs.
// Gate GEMM and proj GEMM run sequentially sharing one 64-VGPR accumulator;
// gate passes through a bf16 LDS tile (keeps LDS <= 136KB -> 2 blocks/WGP).
// ---------------------------------------------------------------------------
__global__ void k_proj(const float* __restrict__ pair, const float* __restrict__ mask,
                       const __bf16* __restrict__ wts,
                       const float* __restrict__ lp_b, const float* __restrict__ lp_g,
                       const float* __restrict__ lp_be, const float* __restrict__ lg_b,
                       const float* __restrict__ rp_b, const float* __restrict__ rp_g,
                       const float* __restrict__ rp_be, const float* __restrict__ rg_b,
                       __bf16* __restrict__ leftcm, __bf16* __restrict__ rightcm) {
    extern __shared__ float sm[];
    float*  tile  = sm;                      // [128][129] fp32 proj tile   (66,048 B)
    float*  sMu   = sm + 128 * 129;          // [128]
    float*  sRs   = sMu + 128;               // [128]
    __bf16* gtb   = (__bf16*)(sRs + 128);    // [128][132] bf16 gate tile   (33,792 B)
    __bf16* apair = gtb + 128 * 132;         // [128][136] bf16 activations (34,816 B)

    const int  tid  = threadIdx.x;
    const int  lane = tid & 31;
    const int  w    = tid >> 5;
    const int  hi8  = (lane >> 4) << 3;
    const long p0   = (long)blockIdx.x * 128;

    // ---- stage pair activations -> LDS bf16 (one conversion per element) ----
#pragma unroll 4
    for (int e = 0; e < 32; ++e) {
        int idx = e * 256 + tid;                   // 8192 float2 pairs
        int p = idx >> 6, c2 = (idx & 63) * 2;
        const float* src = pair + (p0 + p) * (long)C + c2;
        unsigned pk = pack2(f2bf(src[0]), f2bf(src[1]));
        *reinterpret_cast<unsigned*>(apair + p * 136 + c2) = pk;
    }
    __syncthreads();

    // per-lane A pointer base (row = w*16 + lane&15, K chunk select by lane>>4)
    const __bf16* aBase = apair + (w * 16 + (lane & 15)) * 136 + hi8;

    for (int pass = 0; pass < 2; ++pass) {
        const __bf16* wP  = wts + (pass ? 1 : 0) * 16384;
        const __bf16* wG  = wts + (pass ? 3 : 2) * 16384;
        const float*  bP  = pass ? rp_b  : lp_b;
        const float*  bG  = pass ? rg_b  : lg_b;
        const float*  gam = pass ? rp_g  : lp_g;
        const float*  bet = pass ? rp_be : lp_be;
        __bf16*       dst = pass ? rightcm : leftcm;

        v8f acc[8];

        // ---- phase G: gate GEMM -> sigmoid -> LDS gate tile (bf16) ----
#pragma unroll
        for (int t = 0; t < 8; ++t) acc[t] = zero8();
        gemm128_lds_a(aBase, wG, lane, acc);
#pragma unroll
        for (int nt = 0; nt < 8; ++nt) {
            int   c   = nt * 16 + (lane & 15);
            float gbv = bG[c];
#pragma unroll
            for (int r = 0; r < 8; ++r) {
                int row = w * 16 + r + hi8;
                gtb[row * 132 + c] = f2bf(sigmoidf(acc[nt][r] + gbv));
            }
        }

        // ---- phase P: proj GEMM (+bias) -> LDS proj tile ----
#pragma unroll
        for (int t = 0; t < 8; ++t) acc[t] = zero8();
        gemm128_lds_a(aBase, wP, lane, acc);
#pragma unroll
        for (int nt = 0; nt < 8; ++nt) {
            int   c  = nt * 16 + (lane & 15);
            float bv = bP[c];
#pragma unroll
            for (int r = 0; r < 8; ++r) {
                int row = w * 16 + r + hi8;
                tile[row * 129 + c] = acc[nt][r] + bv;
            }
        }
        __syncthreads();

        // ---- LayerNorm stats (one thread per p-row) ----
        if (tid < 128) {
            const float* row = tile + tid * 129;
            float s = 0.f;
            for (int c2 = 0; c2 < C; ++c2) s += row[c2];
            float mu = s * (1.0f / 128.0f);
            float v = 0.f;
            for (int c2 = 0; c2 < C; ++c2) { float d = row[c2] - mu; v += d * d; }
            sMu[tid] = mu;
            sRs[tid] = rsqrtf(v * (1.0f / 128.0f) + 1e-5f);
        }
        __syncthreads();

        // ---- normalize * gamma + beta, * mask2d, * gate ----
        float mk[8], mu8[8], rs8[8];
#pragma unroll
        for (int r = 0; r < 8; ++r) {
            int  row = w * 16 + r + hi8;
            long pg  = p0 + row;
            int  i   = (int)(pg / LSEQ);
            int  kk  = (int)(pg - (long)i * LSEQ);
            mk[r]  = mask[i] * mask[kk];
            mu8[r] = sMu[row];
            rs8[r] = sRs[row];
        }
#pragma unroll
        for (int nt = 0; nt < 8; ++nt) {
            int   c  = nt * 16 + (lane & 15);
            float gv = gam[c], bev = bet[c];
#pragma unroll
            for (int r = 0; r < 8; ++r) {
                int   row = w * 16 + r + hi8;
                float x   = tile[row * 129 + c];
                tile[row * 129 + c] =
                    ((x - mu8[r]) * rs8[r] * gv + bev) * mk[r] * bf2f(gtb[row * 132 + c]);
            }
        }
        __syncthreads();

        // ---- transposed coalesced write: dst[c][p0 + p] as packed 2xbf16 ----
#pragma unroll 4
        for (int it = 0; it < 32; ++it) {
            int c  = it * 4 + (tid >> 6);
            int p2 = (tid & 63) * 2;
            unsigned pk = pack2(f2bf(tile[p2 * 129 + c]),
                                f2bf(tile[(p2 + 1) * 129 + c]));
            *reinterpret_cast<unsigned*>(dst + (size_t)c * P + p0 + p2) = pk;
        }
        __syncthreads();
    }
}

// ---------------------------------------------------------------------------
// Kernel 2: per-channel einsum  out[c][i][j] = sum_k L[c][i][k] * R[c][j][k]
// 128 batched 384x384x384 bf16 GEMMs; both operands (75.5 MB) L2-resident.
// Block: 256 threads (8 waves, 4x2), tile 128i x 128j per channel.
// All 6 fragments of a K-step are loaded before the 8 WMMAs.
// ---------------------------------------------------------------------------
__global__ void k_einsum(const __bf16* __restrict__ leftcm,
                         const __bf16* __restrict__ rightcm,
                         float* __restrict__ outT) {
    const int tid  = threadIdx.x;
    const int lane = tid & 31;
    const int w    = tid >> 5;
    const int wi   = w >> 1;                 // 0..3
    const int wj   = w & 1;                  // 0..1
    const int c    = blockIdx.z;
    const int i0   = blockIdx.x * 128 + wi * 32;   // wave: 32 rows
    const int j0   = blockIdx.y * 128 + wj * 64;   // wave: 64 cols

    const __bf16* lb = leftcm  + (size_t)c * P;
    const __bf16* rb = rightcm + (size_t)c * P;

    v8f acc[2][4];
#pragma unroll
    for (int mt = 0; mt < 2; ++mt)
#pragma unroll
        for (int nt = 0; nt < 4; ++nt) acc[mt][nt] = zero8();

#pragma unroll 2
    for (int k0 = 0; k0 < LSEQ; k0 += 32) {
        if (k0 + 32 < LSEQ) {   // hint next K-step into cache
            __builtin_prefetch(lb + (size_t)(i0 + lane) * LSEQ + k0 + 32, 0, 0);
            __builtin_prefetch(rb + (size_t)(j0 + lane) * LSEQ + k0 + 32, 0, 0);
        }
        v16bf a0 = frag_bf16(lb, LSEQ, i0,      k0, lane);
        v16bf a1 = frag_bf16(lb, LSEQ, i0 + 16, k0, lane);
        v16bf b[4];
#pragma unroll
        for (int nt = 0; nt < 4; ++nt)
            b[nt] = frag_bf16(rb, LSEQ, j0 + nt * 16, k0, lane);
#pragma unroll
        for (int nt = 0; nt < 4; ++nt) {
            acc[0][nt] = wmma_bf16(a0, b[nt], acc[0][nt]);
            acc[1][nt] = wmma_bf16(a1, b[nt], acc[1][nt]);
        }
    }

    float* ob = outT + (size_t)c * P;
#pragma unroll
    for (int mt = 0; mt < 2; ++mt)
#pragma unroll
        for (int nt = 0; nt < 4; ++nt) {
#pragma unroll
            for (int r = 0; r < 8; ++r) {
                int m = i0 + mt * 16 + r + ((lane >> 4) << 3);
                int n = j0 + nt * 16 + (lane & 15);
                ob[(size_t)m * LSEQ + n] = acc[mt][nt][r];
            }
        }
}

// ---------------------------------------------------------------------------
// Kernel 3: channel-LayerNorm + out linear (WMMA) * sigmoid(out-gate (WMMA)).
// Block: 256 threads, 128 positions x 128 channels.
// Gate GEMM runs first and parks its sigmoid output in the f32 LDS tile
// (free after the bf16 LN'd tile is built), halving live accumulators.
// ---------------------------------------------------------------------------
__global__ void k_out(const float* __restrict__ outT, const float* __restrict__ pair,
                      const __bf16* __restrict__ wts,
                      const float* __restrict__ ol_g, const float* __restrict__ ol_be,
                      const float* __restrict__ ob_v, const float* __restrict__ og_b,
                      float* __restrict__ out) {
    extern __shared__ char smraw[];
    float*  tile = (float*)smraw;              // [128][129] fp32 (reused for gate)
    float*  sMu  = tile + 128 * 129;
    float*  sRs  = sMu + 128;
    __bf16* bt   = (__bf16*)(sRs + 128);       // [128][136] bf16 (16B-aligned rows)

    const int  tid  = threadIdx.x;
    const int  lane = tid & 31;
    const int  w    = tid >> 5;
    const int  hi8  = (lane >> 4) << 3;
    const long p0   = (long)blockIdx.x * 128;

    // ---- stage outT [c][p] -> LDS [p][c] ----
#pragma unroll 8
    for (int e = 0; e < 64; ++e) {
        int idx = e * 256 + tid;
        int c = idx >> 7, p = idx & 127;
        tile[p * 129 + c] = outT[(size_t)c * P + p0 + p];
    }
    __syncthreads();

    if (tid < 128) {
        const float* row = tile + tid * 129;
        float s = 0.f;
        for (int c2 = 0; c2 < C; ++c2) s += row[c2];
        float mu = s * (1.0f / 128.0f);
        float v = 0.f;
        for (int c2 = 0; c2 < C; ++c2) { float d = row[c2] - mu; v += d * d; }
        sMu[tid] = mu;
        sRs[tid] = rsqrtf(v * (1.0f / 128.0f) + 1e-5f);
    }
    __syncthreads();

    // ---- normalized bf16 A-matrix for the output GEMM ----
#pragma unroll 8
    for (int e = 0; e < 64; ++e) {
        int idx = e * 256 + tid;
        int c = idx >> 7, p = idx & 127;
        float x = tile[p * 129 + c];
        bt[p * 136 + c] = f2bf((x - sMu[p]) * sRs[p] * ol_g[c] + ol_be[c]);
    }
    __syncthreads();                           // tile is now free

    const __bf16* wOG = wts + 4 * 16384;
    const __bf16* wOW = wts + 5 * 16384;

    v8f acc[8];

    // ---- phase gate: sigmoid(pair @ og^T + og_b) -> LDS tile ----
#pragma unroll
    for (int t = 0; t < 8; ++t) acc[t] = zero8();
#pragma unroll
    for (int ks = 0; ks < 4; ++ks) {
        v16bf apr = frag_f32(pair, C, p0 + w * 16, ks * 32, lane);
        v16bf b[8];
#pragma unroll
        for (int nt = 0; nt < 8; ++nt)
            b[nt] = frag_bf16(wOG, C, nt * 16, ks * 32, lane);
#pragma unroll
        for (int nt = 0; nt < 8; ++nt)
            acc[nt] = wmma_bf16(apr, b[nt], acc[nt]);
    }
#pragma unroll
    for (int nt = 0; nt < 8; ++nt) {
        int   c    = nt * 16 + (lane & 15);
        float ogbv = og_b[c];
#pragma unroll
        for (int r = 0; r < 8; ++r) {
            int row = w * 16 + r + hi8;
            tile[row * 129 + c] = sigmoidf(acc[nt][r] + ogbv);
        }
    }

    // ---- phase out: LN'd @ ow^T + ob ----
    const __bf16* aBase = bt + (w * 16 + (lane & 15)) * 136 + hi8;
#pragma unroll
    for (int t = 0; t < 8; ++t) acc[t] = zero8();
    gemm128_lds_a(aBase, wOW, lane, acc);

    // ---- epilogue: (acc + ob) * gate ----
#pragma unroll
    for (int nt = 0; nt < 8; ++nt) {
        int   c   = nt * 16 + (lane & 15);
        float obv = ob_v[c];
#pragma unroll
        for (int r = 0; r < 8; ++r) {
            int row = w * 16 + r + hi8;
            out[(p0 + row) * C + c] = (acc[nt][r] + obv) * tile[row * 129 + c];
        }
    }
}

// ---------------------------------------------------------------------------
// Launcher
// ---------------------------------------------------------------------------
extern "C" void kernel_launch(void* const* d_in, const int* in_sizes, int n_in,
                              void* d_out, int out_size, void* d_ws, size_t ws_size,
                              hipStream_t stream) {
    (void)in_sizes; (void)n_in; (void)out_size; (void)ws_size;

    const float* pair   = (const float*)d_in[0];
    const float* rmask  = (const float*)d_in[1];
    const float* lp_w   = (const float*)d_in[2];
    const float* lp_b   = (const float*)d_in[3];
    const float* lp_g   = (const float*)d_in[4];
    const float* lp_be  = (const float*)d_in[5];
    const float* rp_w   = (const float*)d_in[6];
    const float* rp_b   = (const float*)d_in[7];
    const float* rp_g   = (const float*)d_in[8];
    const float* rp_be  = (const float*)d_in[9];
    const float* lg_w   = (const float*)d_in[10];
    const float* lg_b   = (const float*)d_in[11];
    const float* rg_w   = (const float*)d_in[12];
    const float* rg_b   = (const float*)d_in[13];
    const float* og_w   = (const float*)d_in[14];
    const float* og_b   = (const float*)d_in[15];
    const float* ol_g   = (const float*)d_in[16];
    const float* ol_be  = (const float*)d_in[17];
    const float* ow     = (const float*)d_in[18];
    const float* ob     = (const float*)d_in[19];
    float*       outp   = (float*)d_out;

    char* ws = (char*)d_ws;
    const size_t bfPlane = (size_t)C * (size_t)P * 2;          // 37,748,736 B
    __bf16* wts     = (__bf16*)ws;                              // 192 KB used
    __bf16* leftcm  = (__bf16*)(ws + 256 * 1024);
    __bf16* rightcm = (__bf16*)(ws + 256 * 1024 + bfPlane);
    float*  outT    = (float*) (ws + 256 * 1024 + 2 * bfPlane); // 75.5 MB

    // k_proj: f32 tile + stats + bf16 gate tile + bf16 activation tile
    const unsigned smProj = (128 * 129 + 256) * 4 + 128 * 132 * 2 + 128 * 136 * 2; // 135,680 B
    // k_out: f32 tile + stats + bf16 tile
    const unsigned smOut  = (128 * 129 + 256) * 4 + 128 * 136 * 2;                 // 101,888 B
    (void)hipFuncSetAttribute((const void*)k_proj,
                              hipFuncAttributeMaxDynamicSharedMemorySize, smProj);
    (void)hipFuncSetAttribute((const void*)k_out,
                              hipFuncAttributeMaxDynamicSharedMemorySize, smOut);

    k_cvt_w<<<dim3(6 * 16384 / 256), dim3(256), 0, stream>>>(
        lp_w, rp_w, lg_w, rg_w, og_w, ow, wts);

    k_proj<<<dim3((unsigned)(P / 128)), dim3(256), smProj, stream>>>(
        pair, rmask, wts,
        lp_b, lp_g, lp_be, lg_b,
        rp_b, rp_g, rp_be, rg_b,
        leftcm, rightcm);

    k_einsum<<<dim3(3, 3, 128), dim3(256), 0, stream>>>(leftcm, rightcm, outT);

    k_out<<<dim3((unsigned)(P / 128)), dim3(256), smOut, stream>>>(
        outT, pair, wts, ol_g, ol_be, ob, og_b, outp);
}
